// QLSTMCell_63153199121065
// MI455X (gfx1250) — matrix operand order repaired and checked by reading, not compiled, not run on hardware
//
#include <hip/hip_runtime.h>

// ---------------------------------------------------------------------------
// QLSTMCell fused kernel for gfx1250 (MI455X)
//   pass 1: fp32 -> f16 conversion of X, Hid, Wih, Whh into workspace
//   pass 2: dual GEMM (gi, gh) via v_wmma_f32_16x16x32_f16, with TDM
//           (tensor_load_to_lds) double-buffered LDS staging + fused
//           quantized LSTM gate epilogue
// ---------------------------------------------------------------------------

typedef __attribute__((ext_vector_type(16))) _Float16 v16h;
typedef __attribute__((ext_vector_type(8)))  _Float16 v8h;
typedef __attribute__((ext_vector_type(4)))  _Float16 v4h;
typedef __attribute__((ext_vector_type(8)))  float    v8f;
typedef __attribute__((ext_vector_type(4)))  unsigned int v4u;
typedef __attribute__((ext_vector_type(8)))  int      v8i;
typedef __attribute__((ext_vector_type(4)))  int      v4i;

#define B_DIM 4096
#define I_DIM 512
#define H_DIM 512

// LDS stage layout (bytes). Rows are 64B of data padded to 80B stride:
// 16B-granular reads then hit 16 distinct banks (stride 4) -> conflict-free.
#define ROWB   80
#define ST_X   0                       // 32 rows
#define ST_H   (ST_X + 32 * ROWB)      // 32 rows
#define ST_WI  (ST_H + 32 * ROWB)      // 256 rows (4 gates x 64 h)
#define ST_WH  (ST_WI + 256 * ROWB)    // 256 rows
#define STAGE  (ST_WH + 256 * ROWB)    // 46080 B per stage, x2 = 90 KB

#if defined(__has_builtin)
#  if __has_builtin(__builtin_amdgcn_tensor_load_to_lds)
#    define HAVE_TDM 1
#  endif
#endif
#ifndef HAVE_TDM
#  define HAVE_TDM 0
#endif

// ---------------- quantization helpers (match reference semantics) ----------

static __device__ __forceinline__ float quant14(float x) {
    return floorf(x * 16384.0f + 0.5f) * (1.0f / 16384.0f);
}
static __device__ __forceinline__ float quant15(float x) {
    return floorf(x * 32768.0f + 0.5f) * (1.0f / 32768.0f);
}
static __device__ __forceinline__ float qin27(float x) {
    float xq = floorf(x * 134217728.0f + 0.5f);
    xq = fminf(fmaxf(xq, -2147483648.0f), 2147483647.0f);
    return xq * (1.0f / 134217728.0f);
}
static __device__ __forceinline__ float qsigmoid_f(float x) {
    float xf = qin27(x);
    float s  = 1.0f / (1.0f + __expf(-xf));
    return floorf(s * 32768.0f + 0.5f) * (1.0f / 32768.0f);
}
static __device__ __forceinline__ float qtanh_f(float x) {
    float xf = qin27(x);
    float t  = tanhf(xf);
    return floorf(t * 32768.0f + 0.5f) * (1.0f / 32768.0f);
}

// ---------------- fragment load from LDS (two 16B chunks -> v16h) -----------

static __device__ __forceinline__ v16h ld2(const char* p, int deltaB) {
    v8h lo = *(const v8h*)(p);
    v8h hi = *(const v8h*)(p + deltaB);
    return __builtin_shufflevector(lo, hi,
             0,1,2,3,4,5,6,7,8,9,10,11,12,13,14,15);
}

static __device__ __forceinline__ v8f wmma_f16(v16h a, v16h b, v8f c) {
    return __builtin_amdgcn_wmma_f32_16x16x32_f16(
        false, a, false, b, (short)0, c, false, false);
}

static __device__ __forceinline__ uint32_t lds_addr_of(const void* p) {
    return (uint32_t)(uintptr_t)(__attribute__((address_space(3))) const void*)p;
}

// ---------------- TDM tile load ---------------------------------------------
// Builds a Tensor DMA Descriptor (D#) per CDNA5 ISA and
// issues TENSOR_LOAD_TO_LDS.  data_size=2B; LDS padding: every 16 DWORDs
// (64B row) insert 4 DWORDs (16B) -> 80B row stride.
#if HAVE_TDM
static __device__ __forceinline__ void tdm_tile_load(
    const _Float16* gsrc, uint32_t lds_byte_addr,
    uint32_t tile_d0, uint32_t tile_d1, uint32_t tile_d2,
    uint32_t tensor_d1, uint32_t stride0, uint32_t stride1)
{
    const uint64_t ga = (uint64_t)(uintptr_t)gsrc;
    v4u g0;
    g0[0] = 1u;                                   // count=1 (valid user D#)
    g0[1] = lds_byte_addr;                        // lds_addr
    g0[2] = (uint32_t)ga;                         // global_addr[31:0]
    g0[3] = (uint32_t)(ga >> 32) | 0x80000000u;   // global_addr[56:32] | type=2

    // w0: data_size=1(2B)<<16 | pad_enable<<20 | pad_interval=3<<22 | pad_amount=3<<25
    const uint32_t w0 = (1u << 16) | (1u << 20) | (3u << 22) | (3u << 25);
    v8i g1;
    g1[0] = (int)w0;
    g1[1] = (int)(512u << 16);                    // tensor_dim0 = 512 (lo16)
    g1[2] = (int)(tensor_d1 << 16);               // dim0 hi=0 | tensor_dim1 lo16
    g1[3] = (int)(tile_d0 << 16);                 // dim1 hi=0 | tile_dim0
    g1[4] = (int)(tile_d1 | (tile_d2 << 16));     // tile_dim1 | tile_dim2
    g1[5] = (int)stride0;                         // tensor_dim0_stride lo32
    g1[6] = (int)((stride1 & 0xFFFFu) << 16);     // d0str hi16=0 | d1str lo16
    g1[7] = (int)(stride1 >> 16);                 // d1str bits 47:16
    v4i g2 = { (int)tile_d2, 0, 0, 0 };           // tensor_dim2 (0 for 2D)
    v4i g3 = { 0, 0, 0, 0 };
#if __clang_major__ >= 23
    v8i g4 = { 0, 0, 0, 0, 0, 0, 0, 0 };
    __builtin_amdgcn_tensor_load_to_lds(g0, g1, g2, g3, g4, 0);
#else
    __builtin_amdgcn_tensor_load_to_lds(g0, g1, g2, g3, 0);
#endif
}
#endif

// ---------------- pass 1: fp32 -> f16 conversion ----------------------------

__global__ void __launch_bounds__(256)
cvt_f32_to_f16(const float* __restrict__ src, _Float16* __restrict__ dst, int n4) {
    int i = blockIdx.x * blockDim.x + threadIdx.x;
    if (i >= n4) return;
    float4 v = ((const float4*)src)[i];
    v4h o;
    o[0] = (_Float16)v.x;
    o[1] = (_Float16)v.y;
    o[2] = (_Float16)v.z;
    o[3] = (_Float16)v.w;
    ((v4h*)dst)[i] = o;
}

// ---------------- pass 2: fused dual-GEMM + quantized LSTM gates -------------
//
// Workgroup: 256 threads = 8 waves, 2(M) x 4(N-of-H). Block tile: 32 rows x
// 64 h-cols x all 4 gates.  K pipeline: 16 steps of K=32, double-buffered LDS
// stages filled by TDM (wave 0), compute reads via ds_load_b128 only.

extern __shared__ char smem_raw[];

__global__ void __launch_bounds__(256)
qlstm_fused(const _Float16* __restrict__ Xh,
            const _Float16* __restrict__ Hh,
            const _Float16* __restrict__ Wih,
            const _Float16* __restrict__ Whh,
            const float*    __restrict__ bias_ih,
            const float*    __restrict__ bias_hh,
            const float*    __restrict__ cx,
            float*          __restrict__ out_hy,
            float*          __restrict__ out_cy) {
    const int lane = threadIdx.x & 31;
    const int w    = threadIdx.x >> 5;
    const int mw   = w & 1;
    const int nw   = w >> 1;
    const int lr   = lane & 15;
    const int lh   = lane >> 4;

    const int m0blk = blockIdx.x * 32;
    const int h0blk = blockIdx.y * 64;
    const int m0 = m0blk + mw * 16;
    const int h0 = h0blk + nw * 16;

#if HAVE_TDM
    const uint32_t lds0 = lds_addr_of(smem_raw);
#endif

    v8f accGi[4] = { {}, {}, {}, {} };
    v8f accGh[4] = { {}, {}, {}, {} };

    // ------------ stage fill ------------
#if HAVE_TDM
#define FILL_STAGE(bufsel, k0)                                                 \
    do { if (w == 0) {                                                         \
        const uint32_t lb_ = lds0 + (uint32_t)(bufsel) * STAGE;                \
        tdm_tile_load(Xh  + (size_t)m0blk * I_DIM + (k0), lb_ + ST_X,          \
                      32, 32, 0, 4096, 512, 0);                                \
        tdm_tile_load(Hh  + (size_t)m0blk * H_DIM + (k0), lb_ + ST_H,          \
                      32, 32, 0, 4096, 512, 0);                                \
        tdm_tile_load(Wih + (size_t)h0blk * I_DIM + (k0), lb_ + ST_WI,         \
                      32, 64, 4, 2048, 512, 262144u);                          \
        tdm_tile_load(Whh + (size_t)h0blk * H_DIM + (k0), lb_ + ST_WH,         \
                      32, 64, 4, 2048, 512, 262144u);                          \
    } } while (0)
#define WAIT_FILL() do { if (w == 0) __builtin_amdgcn_s_wait_tensorcnt((short)0); } while (0)
#else
    // Fallback: cooperative register-staged copy (256 threads, 9 x 16B each).
#define FILL_STAGE(bufsel, k0)                                                 \
    do {                                                                       \
        char* db_ = smem_raw + (size_t)(bufsel) * STAGE;                       \
        const int t_ = threadIdx.x;                                            \
        const int kk_ = (k0);                                                  \
        { int r_ = (t_ & 127) >> 2, c_ = t_ & 3;                               \
          const _Float16* srcA_ =                                              \
              ((t_ < 128) ? Xh + (size_t)(m0blk + r_) * I_DIM                  \
                          : Hh + (size_t)(m0blk + r_) * H_DIM) + kk_ + c_ * 8; \
          *(v8h*)(db_ + ((t_ < 128) ? ST_X : ST_H) + r_ * ROWB + c_ * 16) =    \
              *(const v8h*)srcA_; }                                            \
        { int gg_ = t_ >> 6, hr_ = t_ & 63;                                    \
          const _Float16* si_ =                                                \
              Wih + (size_t)(gg_ * 512 + h0blk + hr_) * I_DIM + kk_;           \
          const _Float16* sh_ =                                                \
              Whh + (size_t)(gg_ * 512 + h0blk + hr_) * H_DIM + kk_;           \
          _Pragma("unroll")                                                    \
          for (int c_ = 0; c_ < 4; ++c_) {                                     \
              *(v8h*)(db_ + ST_WI + t_ * ROWB + c_ * 16) =                     \
                  *(const v8h*)(si_ + c_ * 8);                                 \
              *(v8h*)(db_ + ST_WH + t_ * ROWB + c_ * 16) =                     \
                  *(const v8h*)(sh_ + c_ * 8);                                 \
          } }                                                                  \
    } while (0)
#define WAIT_FILL() do { } while (0)
#endif

    // ------------ pipeline prologue ------------
    FILL_STAGE(0, 0);
    WAIT_FILL();
    __syncthreads();

    const int arow = (mw * 16 + lr) * ROWB;   // A-tile row byte offset
    const int ac   = lh * 16;                 // A K-half chunk offset
    const int bc   = lh * 16;                 // B K-half chunk offset

    // ------------ 16 K-steps, double-buffered ------------
#pragma unroll
    for (int s = 0; s < 16; ++s) {
        const int cur = s & 1;
        if (s < 15) FILL_STAGE(cur ^ 1, 32 * (s + 1));

        const char* cb = smem_raw + (size_t)cur * STAGE;

        // Issue ALL fragment ds_loads first (in-order LDS returns ->
        // graduated s_wait_dscnt), then stream the 8 WMMAs.
        const v16h ax = ld2(cb + ST_X + arow + ac, 32);
        const v16h ah = ld2(cb + ST_H + arow + ac, 32);
        v16h bwi[4], bwh[4];
#pragma unroll
        for (int g = 0; g < 4; ++g) {
            const int brow = (g * 64 + nw * 16 + lr) * ROWB;
            bwi[g] = ld2(cb + ST_WI + brow + bc, 32);
            bwh[g] = ld2(cb + ST_WH + brow + bc, 32);
        }
#pragma unroll
        for (int g = 0; g < 4; ++g) {
            accGi[g] = wmma_f16(ax, bwi[g], accGi[g]);
            accGh[g] = wmma_f16(ah, bwh[g], accGh[g]);
        }

        if (s < 15) {
            WAIT_FILL();
            __syncthreads();
        }
    }

    // ------------ fused quantized-gate epilogue ------------
    const int h = h0 + lr;
    float bih[4], bhh[4];
#pragma unroll
    for (int g = 0; g < 4; ++g) {
        bih[g] = bias_ih[g * 512 + h];
        bhh[g] = bias_hh[g * 512 + h];
    }

#pragma unroll
    for (int r = 0; r < 8; ++r) {
        const int m = m0 + r + lh * 8;
        const size_t idx = (size_t)m * H_DIM + h;

        float gate[4];
#pragma unroll
        for (int g = 0; g < 4; ++g) {
            float gi = quant14(accGi[g][r] + bih[g]);
            float gh = quant14(accGh[g][r] + bhh[g]);
            gate[g] = gi + gh;
        }

        const float ing = qsigmoid_f(gate[0]);
        const float fg  = qsigmoid_f(gate[1]);
        const float cg  = qtanh_f(gate[2]);
        const float og  = qsigmoid_f(gate[3]);

        const float qcx = quant15(cx[idx]);
        const float cyv = qcx * fg + ing * cg;
        const float hyv = qtanh_f(cyv) * og;

        out_hy[idx] = hyv;
        out_cy[idx] = quant15(cyv);
    }
}

// ---------------------------------------------------------------------------

extern "C" void kernel_launch(void* const* d_in, const int* in_sizes, int n_in,
                              void* d_out, int out_size, void* d_ws, size_t ws_size,
                              hipStream_t stream) {
    const float* input_x   = (const float*)d_in[0];
    const float* hidden    = (const float*)d_in[1];
    const float* cx        = (const float*)d_in[2];
    const float* weight_ih = (const float*)d_in[3];
    const float* weight_hh = (const float*)d_in[4];
    const float* bias_ih   = (const float*)d_in[5];
    const float* bias_hh   = (const float*)d_in[6];

    const size_t nX = (size_t)B_DIM * I_DIM;
    const size_t nW = (size_t)4 * H_DIM * I_DIM;

    _Float16* Xh  = (_Float16*)d_ws;
    _Float16* Hh  = Xh + nX;
    _Float16* Wih = Hh + nX;
    _Float16* Whh = Wih + nW;

    {
        int n4;
        n4 = (int)(nX / 4);
        cvt_f32_to_f16<<<(n4 + 255) / 256, 256, 0, stream>>>(input_x, Xh, n4);
        cvt_f32_to_f16<<<(n4 + 255) / 256, 256, 0, stream>>>(hidden,  Hh, n4);
        n4 = (int)(nW / 4);
        cvt_f32_to_f16<<<(n4 + 255) / 256, 256, 0, stream>>>(weight_ih, Wih, n4);
        cvt_f32_to_f16<<<(n4 + 255) / 256, 256, 0, stream>>>(weight_hh, Whh, n4);
    }

    float* out_hy = (float*)d_out;
    float* out_cy = out_hy + (size_t)B_DIM * H_DIM;

    (void)hipFuncSetAttribute((const void*)qlstm_fused,
                              hipFuncAttributeMaxDynamicSharedMemorySize,
                              2 * STAGE);

    dim3 grid(B_DIM / 32, H_DIM / 64);  // 128 x 8 workgroups
    qlstm_fused<<<grid, 256, 2 * STAGE, stream>>>(Xh, Hh, Wih, Whh,
                                                  bias_ih, bias_hh, cx,
                                                  out_hy, out_cy);
}
